// SelfAttentionBlock_37726992728938
// MI455X (gfx1250) — matrix operand order repaired and checked by reading, not compiled, not run on hardware
//
#include <hip/hip_runtime.h>
#include <hip/hip_bf16.h>
#include <math.h>

#define EMBED 2048
#define HEADS 16
#define HEAD_DIM 128
#define SEQ 2048

typedef __attribute__((ext_vector_type(16))) __bf16 v16bf;
typedef __attribute__((ext_vector_type(8)))  __bf16 v8bf;
typedef __attribute__((ext_vector_type(8)))  float  v8f;

__device__ __forceinline__ v8f zero_v8f() {
  v8f v;
#pragma unroll
  for (int i = 0; i < 8; ++i) v[i] = 0.0f;
  return v;
}

// A fragment: 16(M) x 32(K) bf16 from row-major memory (ld elements per row).
// Lane L<16: M=L, K = {0..7, 16..23}; lane L>=16: M=L-16, K = {8..15, 24..31}.
__device__ __forceinline__ v16bf load_frag_a(const __bf16* base, int ld) {
  const int lane = threadIdx.x & 31;
  const int row  = lane & 15;
  const int koff = (lane >> 4) << 3;
  const __bf16* p = base + (size_t)row * ld + koff;
  v8bf lo = *(const v8bf*)(p);
  v8bf hi = *(const v8bf*)(p + 16);
  v16bf r;
#pragma unroll
  for (int i = 0; i < 8; ++i) { r[i] = lo[i]; r[i + 8] = hi[i]; }
  return r;
}

// B fragment: 32(K) x 16(N) bf16. Memory holds B^T row-major: row n is 32
// contiguous K values (ld elements between n rows). Lane L: N = L&15,
// K = {0..7, 16..23} for L<16, {8..15, 24..31} for L>=16.
__device__ __forceinline__ v16bf load_frag_b(const __bf16* base, int ld) {
  const int lane = threadIdx.x & 31;
  const int n    = lane & 15;
  const int koff = (lane >> 4) << 3;
  const __bf16* p = base + (size_t)n * ld + koff;
  v8bf lo = *(const v8bf*)(p);
  v8bf hi = *(const v8bf*)(p + 16);
  v16bf r;
#pragma unroll
  for (int i = 0; i < 8; ++i) { r[i] = lo[i]; r[i + 8] = hi[i]; }
  return r;
}

// ---------------------------------------------------------------------------
// C = A(MxK) * W^T + bias, W stored [N,K] row-major (einsum bsd,od->bso).
// Block: 256 threads = 8 waves as 2(M) x 4(N); wave tile 32(M) x 64(N):
// 8 WMMA per K-step fed by 6 fragment loads (2 A + 4 B).
// Block tile = 64(M) x 256(N).
// ---------------------------------------------------------------------------
__global__ __launch_bounds__(256) void gemm_bf16_kernel(
    const __bf16* __restrict__ A, const __bf16* __restrict__ W,
    const float* __restrict__ bias, float* __restrict__ C,
    int M, int N, int K) {
  const int lane = threadIdx.x & 31;
  const int wid  = threadIdx.x >> 5;
  const int wm   = wid >> 2;                 // 0..1
  const int wn   = wid & 3;                  // 0..3
  const int m0   = blockIdx.y * 64 + wm * 32;
  const int n0   = blockIdx.x * 256 + wn * 64;

  v8f acc[2][4];
#pragma unroll
  for (int i = 0; i < 2; ++i)
#pragma unroll
    for (int j = 0; j < 4; ++j) acc[i][j] = zero_v8f();

  for (int k0 = 0; k0 < K; k0 += 32) {
    v16bf a[2], b[4];
#pragma unroll
    for (int i = 0; i < 2; ++i)
      a[i] = load_frag_a(A + (size_t)(m0 + i * 16) * K + k0, K);
#pragma unroll
    for (int j = 0; j < 4; ++j)
      b[j] = load_frag_b(W + (size_t)(n0 + j * 16) * K + k0, K);
#pragma unroll
    for (int i = 0; i < 2; ++i)
#pragma unroll
      for (int j = 0; j < 4; ++j)
        acc[i][j] = __builtin_amdgcn_wmma_f32_16x16x32_bf16(
            false, a[i], false, b[j], (short)0, acc[i][j], false, false);
  }

  const int col = lane & 15;
  const int r0  = (lane >> 4) << 3;
#pragma unroll
  for (int i = 0; i < 2; ++i) {
#pragma unroll
    for (int j = 0; j < 4; ++j) {
      const int n = n0 + j * 16 + col;
      const float bv = bias[n];
#pragma unroll
      for (int r = 0; r < 8; ++r) {
        C[(size_t)(m0 + i * 16 + r0 + r) * N + n] = acc[i][j][r] + bv;
      }
    }
  }
}

// ---------------------------------------------------------------------------
// elementwise f32 -> bf16
// ---------------------------------------------------------------------------
__global__ void f32_to_bf16_kernel(const float* __restrict__ in,
                                   __bf16* __restrict__ out, long n) {
  for (long i = (long)blockIdx.x * blockDim.x + threadIdx.x; i < n;
       i += (long)gridDim.x * blockDim.x)
    out[i] = (__bf16)in[i];
}

// ---------------------------------------------------------------------------
// interleaved RoPE on [rows, EMBED] f32 -> bf16 (rows = B*S, head dim 128)
// ---------------------------------------------------------------------------
__global__ void rope_bf16_kernel(const float* __restrict__ in,
                                 __bf16* __restrict__ out, int rows, int S) {
  const float log_base_over_dim = logf(10000.0f) / 64.0f;
  const long total = (long)rows * (EMBED / 2);
  for (long idx = (long)blockIdx.x * blockDim.x + threadIdx.x; idx < total;
       idx += (long)gridDim.x * blockDim.x) {
    const int row = (int)(idx >> 10);       // EMBED/2 = 1024 pairs per row
    const int p   = (int)(idx & 1023);
    const int i   = p & 63;                 // pair index within head
    const int h   = p >> 6;
    const int s   = row & (S - 1);          // S is a power of two
    const float ang = (float)s * __expf(-(float)i * log_base_over_dim);
    const float c = __cosf(ang), sn = __sinf(ang);
    const size_t base = ((size_t)row << 11) + ((size_t)h << 7) + (size_t)(i << 1);
    const float t1 = in[base], t2 = in[base + 1];
    out[base]     = (__bf16)(t1 * c - t2 * sn);
    out[base + 1] = (__bf16)(t1 * sn + t2 * c);
  }
}

// ---------------------------------------------------------------------------
// V: [B*S, EMBED] f32 -> Vt [B, H, D, S] bf16 (transpose for PV B-fragments)
// ---------------------------------------------------------------------------
__global__ void v_transpose_bf16_kernel(const float* __restrict__ in,
                                        __bf16* __restrict__ out, int B, int S) {
  const long total = (long)B * S * EMBED;
  for (long idx = (long)blockIdx.x * blockDim.x + threadIdx.x; idx < total;
       idx += (long)gridDim.x * blockDim.x) {
    const int row  = (int)(idx >> 11);      // b*S + s
    const int colE = (int)(idx & 2047);     // h*128 + d
    const int b    = row / S;
    const int s    = row & (S - 1);
    out[((size_t)b * EMBED + colE) * S + s] = (__bf16)in[idx];
  }
}

// ---------------------------------------------------------------------------
// Flash attention: one wave per 16-row Q tile, online softmax over 32-wide
// causal K blocks. Q,K: [B*S, EMBED] bf16; Vt: [B, H, D, S] bf16;
// O: [B*S, EMBED] bf16.
// ---------------------------------------------------------------------------
__global__ __launch_bounds__(256) void flash_attn_kernel(
    const __bf16* __restrict__ Q, const __bf16* __restrict__ Km,
    const __bf16* __restrict__ Vt, __bf16* __restrict__ O, int B, int S) {
  __shared__ __bf16 Plds[8][16 * 32];
  const int lane = threadIdx.x & 31;
  const int wid  = threadIdx.x >> 5;
  const int widx = blockIdx.x * 8 + wid;
  const int qtiles = S >> 4;
  const int qt = widx % qtiles;
  const int bh = widx / qtiles;
  const int h  = bh % HEADS;
  const int b  = bh / HEADS;
  const int q0 = qt << 4;
  const int r0  = (lane >> 4) << 3;
  const int col = lane & 15;

  const __bf16* qbase = Q + ((size_t)(b * S + q0)) * EMBED + h * HEAD_DIM;
  v16bf qf[4];
#pragma unroll
  for (int c = 0; c < 4; ++c) qf[c] = load_frag_a(qbase + c * 32, EMBED);

  v8f acc[8];
#pragma unroll
  for (int n = 0; n < 8; ++n) acc[n] = zero_v8f();
  float m[8], l[8];
#pragma unroll
  for (int r = 0; r < 8; ++r) { m[r] = -3.0e38f; l[r] = 0.0f; }

  const float scale = 0.088388347648318447f;  // 1/sqrt(128)
  const int kblocks = (q0 + 16 + 31) >> 5;
  __bf16* P = &Plds[wid][0];

  for (int kb = 0; kb < kblocks; ++kb) {
    const int k0 = kb << 5;
    const __bf16* kbase = Km + ((size_t)(b * S + k0)) * EMBED + h * HEAD_DIM;

    // scores: two 16x16 tiles covering k columns [k0..k0+15], [k0+16..k0+31]
    v8f s0 = zero_v8f(), s1 = zero_v8f();
#pragma unroll
    for (int c = 0; c < 4; ++c) {
      v16bf kf0 = load_frag_b(kbase + c * 32, EMBED);
      v16bf kf1 = load_frag_b(kbase + (size_t)16 * EMBED + c * 32, EMBED);
      s0 = __builtin_amdgcn_wmma_f32_16x16x32_bf16(false, qf[c], false, kf0, (short)0, s0, false, false);
      s1 = __builtin_amdgcn_wmma_f32_16x16x32_bf16(false, qf[c], false, kf1, (short)0, s1, false, false);
    }

    // causal mask + block row max
    float bm[8], alpha[8], p0[8], p1[8];
#pragma unroll
    for (int r = 0; r < 8; ++r) {
      const int gq = q0 + r0 + r;
      const float x0 = (k0 + col      <= gq) ? s0[r] * scale : -3.0e38f;
      const float x1 = (k0 + 16 + col <= gq) ? s1[r] * scale : -3.0e38f;
      s0[r] = x0; s1[r] = x1;
      bm[r] = fmaxf(x0, x1);
    }
#pragma unroll
    for (int off = 1; off < 16; off <<= 1) {
#pragma unroll
      for (int r = 0; r < 8; ++r) bm[r] = fmaxf(bm[r], __shfl_xor(bm[r], off, 32));
    }

    // online softmax update
#pragma unroll
    for (int r = 0; r < 8; ++r) {
      const float mn = fmaxf(m[r], bm[r]);
      alpha[r] = __expf(m[r] - mn);
      m[r] = mn;
      p0[r] = __expf(s0[r] - mn);
      p1[r] = __expf(s1[r] - mn);
    }
    float rs[8];
#pragma unroll
    for (int r = 0; r < 8; ++r) rs[r] = p0[r] + p1[r];
#pragma unroll
    for (int off = 1; off < 16; off <<= 1) {
#pragma unroll
      for (int r = 0; r < 8; ++r) rs[r] += __shfl_xor(rs[r], off, 32);
    }
#pragma unroll
    for (int r = 0; r < 8; ++r) l[r] = l[r] * alpha[r] + rs[r];
#pragma unroll
    for (int n = 0; n < 8; ++n)
#pragma unroll
      for (int r = 0; r < 8; ++r) acc[n][r] *= alpha[r];

    // P (C-layout f32) -> LDS bf16 row-major [16 x 32] -> A fragment
#pragma unroll
    for (int r = 0; r < 8; ++r) {
      P[(r0 + r) * 32 + col]      = (__bf16)p0[r];
      P[(r0 + r) * 32 + 16 + col] = (__bf16)p1[r];
    }
    v16bf pa;
    {
      const __bf16* pp = P + (size_t)(lane & 15) * 32 + ((lane >> 4) << 3);
      v8bf lo = *(const v8bf*)(pp);
      v8bf hi = *(const v8bf*)(pp + 16);
#pragma unroll
      for (int i = 0; i < 8; ++i) { pa[i] = lo[i]; pa[i + 8] = hi[i]; }
    }

    // acc += P(16x32) * V(32x128)  — 8 d-tiles
    const __bf16* vtbase = Vt + ((size_t)b * EMBED + h * HEAD_DIM) * S + k0;
#pragma unroll
    for (int n = 0; n < 8; ++n) {
      v16bf vb = load_frag_b(vtbase + (size_t)(n * 16) * S, S);
      acc[n] = __builtin_amdgcn_wmma_f32_16x16x32_bf16(false, pa, false, vb, (short)0, acc[n], false, false);
    }
  }

  // O = acc / l
  __bf16* obase = O + ((size_t)(b * S + q0)) * EMBED + h * HEAD_DIM;
#pragma unroll
  for (int n = 0; n < 8; ++n) {
#pragma unroll
    for (int r = 0; r < 8; ++r) {
      obase[(size_t)(r0 + r) * EMBED + n * 16 + col] = (__bf16)(acc[n][r] / l[r]);
    }
  }
}

// ---------------------------------------------------------------------------
extern "C" void kernel_launch(void* const* d_in, const int* in_sizes, int n_in,
                              void* d_out, int out_size, void* d_ws, size_t ws_size,
                              hipStream_t stream) {
  (void)n_in; (void)out_size; (void)ws_size;
  const float* x  = (const float*)d_in[0];
  const float* wq = (const float*)d_in[1];
  const float* bq = (const float*)d_in[2];
  const float* wk = (const float*)d_in[3];
  const float* bk = (const float*)d_in[4];
  const float* wv = (const float*)d_in[5];
  const float* bv = (const float*)d_in[6];
  const float* wo = (const float*)d_in[7];
  const float* bo = (const float*)d_in[8];

  const int S = SEQ, E = EMBED;
  const int Bb = in_sizes[0] / (S * E);
  const int M  = Bb * S;

  char* w = (char*)d_ws;
  const size_t szME = (size_t)M * E;
  const size_t szEE = (size_t)E * E;
  __bf16* xb   = (__bf16*)w; w += szME * sizeof(__bf16);
  __bf16* wqb  = (__bf16*)w; w += szEE * sizeof(__bf16);
  __bf16* wkb  = (__bf16*)w; w += szEE * sizeof(__bf16);
  __bf16* wvb  = (__bf16*)w; w += szEE * sizeof(__bf16);
  __bf16* wob  = (__bf16*)w; w += szEE * sizeof(__bf16);
  float*  f32s = (float*)w;  w += szME * sizeof(float);
  __bf16* Qb   = (__bf16*)w; w += szME * sizeof(__bf16);
  __bf16* Kb   = (__bf16*)w; w += szME * sizeof(__bf16);
  __bf16* Vt   = (__bf16*)w; w += szME * sizeof(__bf16);
  __bf16* Ob   = (__bf16*)w; w += szME * sizeof(__bf16);

  // 1) convert inputs to bf16
  f32_to_bf16_kernel<<<2048, 256, 0, stream>>>(x,  xb,  (long)szME);
  f32_to_bf16_kernel<<<1024, 256, 0, stream>>>(wq, wqb, (long)szEE);
  f32_to_bf16_kernel<<<1024, 256, 0, stream>>>(wk, wkb, (long)szEE);
  f32_to_bf16_kernel<<<1024, 256, 0, stream>>>(wv, wvb, (long)szEE);
  f32_to_bf16_kernel<<<1024, 256, 0, stream>>>(wo, wob, (long)szEE);

  dim3 ggrid(E / 256, M / 64);

  // 2) Q = x*wq^T + bq ; RoPE -> bf16
  gemm_bf16_kernel<<<ggrid, 256, 0, stream>>>(xb, wqb, bq, f32s, M, E, E);
  rope_bf16_kernel<<<2048, 256, 0, stream>>>(f32s, Qb, M, S);

  // 3) K = x*wk^T + bk ; RoPE -> bf16
  gemm_bf16_kernel<<<ggrid, 256, 0, stream>>>(xb, wkb, bk, f32s, M, E, E);
  rope_bf16_kernel<<<2048, 256, 0, stream>>>(f32s, Kb, M, S);

  // 4) V = x*wv^T + bv ; transpose to [B,H,D,S] bf16
  gemm_bf16_kernel<<<ggrid, 256, 0, stream>>>(xb, wvb, bv, f32s, M, E, E);
  v_transpose_bf16_kernel<<<2048, 256, 0, stream>>>(f32s, Vt, Bb, S);

  // 5) attention
  const int flashBlocks = (Bb * HEADS * (S / 16)) / 8;
  flash_attn_kernel<<<flashBlocks, 256, 0, stream>>>(Qb, Kb, Vt, Ob, Bb, S);

  // 6) out = O*wo^T + bo (f32)
  gemm_bf16_kernel<<<ggrid, 256, 0, stream>>>(Ob, wob, bo, (float*)d_out, M, E, E);
}
